// MonotonicAttention_16432544875122
// MI455X (gfx1250) — matrix operand order repaired and checked by reading, not compiled, not run on hardware
//
#include <hip/hip_runtime.h>
#include <hip/hip_bf16.h>

// ---------------------------------------------------------------------------
// MoChA (monotonic chunkwise attention) forward for MI455X / gfx1250.
// GEMMs: v_wmma_f32_16x16x32_bf16, register-blocked 32x64 per wave, all
// operands laid out so every fragment is two contiguous 16B-per-lane loads.
// Scans: wave32 shfl scans + minimal LDS combines.
// ---------------------------------------------------------------------------

typedef __bf16 bf16;
typedef __attribute__((ext_vector_type(16))) __bf16 v16bf;
typedef __attribute__((ext_vector_type(8)))  float  v8f;

#define Bsz   4
#define QLEN  512
#define KLEN  1024
#define ADIM  512
#define H_MA  4
#define H_CA  2
#define EPSc  1e-6f
#define SCALEc 22.62741699796952f   // sqrt(512)

#define MT 2   // 16-row tiles per wave
#define NT 4   // 16-col tiles per wave

// ---------------- WMMA fragment helpers (ISA 7.12.2 layouts) ----------------

static __device__ __forceinline__ v8f wmma_bf16(v16bf a, v16bf b, v8f c) {
  return __builtin_amdgcn_wmma_f32_16x16x32_bf16(false, a, false, b,
                                                 (short)0, c, false, false);
}

// A-matrix 16x32 bf16, memory row-major [row, k] with leading-dim ld.
// lane: row = lane&15, half = lane>>4; VGPR v: K = 16*(v/4) + 8*half + 2*(v%4).
static __device__ __forceinline__ v16bf frag_rowmajor(const bf16* __restrict__ base,
                                                      int ld, int r0, int k0, int lane) {
  int r = r0 + (lane & 15);
  int half = lane >> 4;
  const bf16* row = base + (size_t)r * ld + k0;
  v16bf f;
#pragma unroll
  for (int v = 0; v < 8; ++v) {
    int k = ((v >> 2) << 4) + (half << 3) + ((v & 3) << 1);
    f[2 * v]     = row[k];
    f[2 * v + 1] = row[k + 1];
  }
  return f;
}

// B-matrix 32x16 bf16 where memory is row-major [n, k] (all B operands are
// stored n-major so each lane reads two contiguous 16-byte runs).
// lane = column n; half = lane>>4; VGPR v: K = 16*(v/4) + 8*half + 2*(v%4).
static __device__ __forceinline__ v16bf frag_b_nmajor(const bf16* __restrict__ base,
                                                      int ld, int k0, int n0, int lane) {
  int n = n0 + (lane & 15);
  int half = lane >> 4;
  const bf16* row = base + (size_t)n * ld + k0;
  v16bf f;
#pragma unroll
  for (int v = 0; v < 8; ++v) {
    int k = ((v >> 2) << 4) + (half << 3) + ((v & 3) << 1);
    f[2 * v]     = row[k];
    f[2 * v + 1] = row[k + 1];
  }
  return f;
}

// ---------------- elementwise conversions ------------------------------------

__global__ void f32_to_bf16_kernel(const float* __restrict__ in,
                                   bf16* __restrict__ out, int n) {
  int i = blockIdx.x * blockDim.x + threadIdx.x;
  if (i < n) out[i] = (bf16)in[i];
}

// in [R,C] f32 -> out [C,R] bf16 (weights: one-time, 256K elements each)
__global__ void f32_to_bf16_T_kernel(const float* __restrict__ in,
                                     bf16* __restrict__ out, int R, int C) {
  int i = blockIdx.x * blockDim.x + threadIdx.x;
  if (i < R * C) {
    int r = i / C, c = i % C;
    out[(size_t)c * R + r] = (bf16)in[i];
  }
}

// ---------------- GEMM with bias: out[M,N] = A[M,K] @ W^T + bias -------------
// W is stored TRANSPOSED [N,K]. One wave computes a 32x64 tile: 2x4 WMMA
// accumulators, 8 WMMAs per 32-wide K step reusing 2 A-frags / 4 B-frags.

template <bool F32OUT, bool TRANSOUT>
__global__ __launch_bounds__(32)
void gemm_bias_kernel(const bf16* __restrict__ A, int K,
                      const bf16* __restrict__ WT, int N, int Mtot,
                      const float* __restrict__ bias,
                      void* __restrict__ out) {
  int lane = threadIdx.x;
  int n0 = blockIdx.x * (16 * NT);
  int m0 = blockIdx.y * (16 * MT);
  v8f acc[MT][NT] = {};
  for (int k0 = 0; k0 < K; k0 += 32) {
    if (k0 + 64 < K) {
      __builtin_prefetch(A + (size_t)(m0 + (lane & 15)) * K + k0 + 64, 0, 3);
      __builtin_prefetch(WT + (size_t)(n0 + (lane & 15)) * K + k0 + 64, 0, 3);
    }
    v16bf a[MT], b[NT];
#pragma unroll
    for (int i = 0; i < MT; ++i)
      a[i] = frag_rowmajor(A, K, m0 + 16 * i, k0, lane);
#pragma unroll
    for (int j = 0; j < NT; ++j)
      b[j] = frag_b_nmajor(WT, K, k0, n0 + 16 * j, lane);
#pragma unroll
    for (int i = 0; i < MT; ++i)
#pragma unroll
      for (int j = 0; j < NT; ++j)
        acc[i][j] = wmma_bf16(a[i], b[j], acc[i][j]);
  }
  int half = lane >> 4;
#pragma unroll
  for (int j = 0; j < NT; ++j) {
    int n = n0 + 16 * j + (lane & 15);
    float bn = bias ? bias[n] : 0.0f;
#pragma unroll
    for (int i = 0; i < MT; ++i)
#pragma unroll
      for (int v = 0; v < 8; ++v) {
        int m = m0 + 16 * i + v + (half << 3);
        float val = acc[i][j][v] + bn;
        if (F32OUT)        ((float*)out)[(size_t)m * N + n] = val;
        else if (TRANSOUT) ((bf16*) out)[(size_t)n * Mtot + m] = (bf16)val;
        else               ((bf16*) out)[(size_t)m * N + n] = (bf16)val;
      }
  }
}

// ---------------- batched per-head energy: e = (q . k^T)*scale + r ----------
// Both operands row-major per head; B fragment = K-projection rows (n-major).

__global__ __launch_bounds__(32)
void energy_kernel(const bf16* __restrict__ Qp, const bf16* __restrict__ Kp,
                   float* __restrict__ E, int H, int dk, float scale,
                   const float* __restrict__ radd) {
  int lane = threadIdx.x;
  int bz = blockIdx.z;
  int b = bz / H, h = bz % H;
  int n0 = blockIdx.x * (16 * NT);   // key index
  int m0 = blockIdx.y * (16 * MT);   // query index
  const bf16* Qb = Qp + (size_t)b * QLEN * ADIM + h * dk;
  const bf16* Kb = Kp + (size_t)b * KLEN * ADIM + h * dk;
  v8f acc[MT][NT] = {};
  for (int k0 = 0; k0 < dk; k0 += 32) {
    v16bf a[MT], bb[NT];
#pragma unroll
    for (int i = 0; i < MT; ++i)
      a[i] = frag_rowmajor(Qb, ADIM, m0 + 16 * i, k0, lane);
#pragma unroll
    for (int j = 0; j < NT; ++j)
      bb[j] = frag_b_nmajor(Kb, ADIM, k0, n0 + 16 * j, lane);
#pragma unroll
    for (int i = 0; i < MT; ++i)
#pragma unroll
      for (int j = 0; j < NT; ++j)
        acc[i][j] = wmma_bf16(a[i], bb[j], acc[i][j]);
  }
  float rv = radd ? radd[0] : 0.0f;
  int half = lane >> 4;
  float* Eb = E + (((size_t)b * H + h) * QLEN) * KLEN;
#pragma unroll
  for (int i = 0; i < MT; ++i)
#pragma unroll
    for (int j = 0; j < NT; ++j) {
      int n = n0 + 16 * j + (lane & 15);
#pragma unroll
      for (int v = 0; v < 8; ++v) {
        int m = m0 + 16 * i + v + (half << 3);
        Eb[(size_t)m * KLEN + n] = acc[i][j][v] * scale + rv;
      }
    }
}

// ---------------- cp = exclusive cumprod(1-p) in log space ------------------
// One 256-thread block per (b,h,q) row of 1024; float4 per thread + LDS scan.

__global__ __launch_bounds__(256)
void cp_kernel(const float* __restrict__ E, float* __restrict__ CP) {
  __shared__ float part[256];
  int row = blockIdx.x;
  const float4* e4 = (const float4*)(E + (size_t)row * KLEN);
  float4*      c4 = (float4*)(CP + (size_t)row * KLEN);
  int tid = threadIdx.x;
  float4 ev = e4[tid];
  float l[4], pre[4], run = 0.0f;
  float es[4] = {ev.x, ev.y, ev.z, ev.w};
#pragma unroll
  for (int j = 0; j < 4; ++j) {
    float p  = 1.0f / (1.0f + __expf(-es[j]));
    float om = fminf(fmaxf(1.0f - p, EPSc), 1.0f);
    l[j] = __logf(om);
    pre[j] = run;
    run += l[j];
  }
  part[tid] = run;
  __syncthreads();
#pragma unroll
  for (int off = 1; off < 256; off <<= 1) {
    float add = (tid >= off) ? part[tid - off] : 0.0f;
    __syncthreads();
    part[tid] += add;
    __syncthreads();
  }
  float base = (tid > 0) ? part[tid - 1] : 0.0f;
  float4 o;
  o.x = __expf(base + pre[0]);
  o.y = __expf(base + pre[1]);
  o.z = __expf(base + pre[2]);
  o.w = __expf(base + pre[3]);
  c4[tid] = o;
}

// ---------------- sequential MoChA recurrence over query steps --------------
// One 1024-thread (32-wave) block per (b,h); thread = key position.
// Per q step: wave32 shfl inclusive scan (no barriers) + 32-partial LDS
// combine -> only 2 barriers per step on the sequential critical path.

__global__ __launch_bounds__(1024)
void alpha_scan_kernel(const float* __restrict__ E, const float* __restrict__ CP,
                       float* __restrict__ ALPHA) {
  __shared__ float wsum[32];
  __shared__ float wbase[32];
  int tid = threadIdx.x;
  int lane = tid & 31;
  int wid = tid >> 5;
  size_t base = (size_t)blockIdx.x * QLEN * KLEN + tid;
  float aw = (tid == 0) ? 1.0f : 0.0f;
  for (int q = 0; q < QLEN; ++q) {
    size_t idx = base + (size_t)q * KLEN;
    float ev  = E[idx];
    float p   = 1.0f / (1.0f + __expf(-ev));
    float cpv = CP[idx];
    float cpc = fminf(fmaxf(cpv, EPSc), 1.0f);
    float v = aw / cpc;
    // intra-wave inclusive scan (wave32)
#pragma unroll
    for (int off = 1; off < 32; off <<= 1) {
      float u = __shfl_up(v, off, 32);
      if (lane >= off) v += u;
    }
    if (lane == 31) wsum[wid] = v;
    __syncthreads();
    if (wid == 0) {
      float s = wsum[lane];
#pragma unroll
      for (int off = 1; off < 32; off <<= 1) {
        float u = __shfl_up(s, off, 32);
        if (lane >= off) s += u;
      }
      wbase[lane] = s - wsum[lane];   // exclusive prefix of wave totals
    }
    __syncthreads();
    float cum = v + wbase[wid];
    float a = p * cpv * cum;
    ALPHA[idx] = a;
    aw = a;
  }
}

// ---------------- chunkwise softmax + beta (fused, per row) -----------------
// One 256-thread block per (b,h_ca,q) row: rowmax, se=max(exp,1e-5),
// denom = movsum(se,3,0); for each h_ma: frac=alpha/denom,
// beta = se * movsum(frac,0,3) -> bf16 [B, 8, Q, K].

__global__ __launch_bounds__(256)
void beta_kernel(const float* __restrict__ ECA, const float* __restrict__ ALPHA,
                 bf16* __restrict__ BETA) {
  __shared__ float se[KLEN];
  __shared__ float fr[KLEN];
  __shared__ float red[256];
  int row = blockIdx.x;                 // (b*H_CA + hca)*QLEN + q
  int q   = row % QLEN;
  int bh  = row / QLEN;
  int b   = bh / H_CA, hca = bh % H_CA;
  int tid = threadIdx.x;

  const float4* e4 = (const float4*)(ECA + (size_t)row * KLEN);
  float4 ev = e4[tid];
  float es[4] = {ev.x, ev.y, ev.z, ev.w};
  float m = fmaxf(fmaxf(es[0], es[1]), fmaxf(es[2], es[3]));
  red[tid] = m;
  __syncthreads();
#pragma unroll
  for (int off = 128; off > 0; off >>= 1) {
    if (tid < off) red[tid] = fmaxf(red[tid], red[tid + off]);
    __syncthreads();
  }
  m = red[0];
#pragma unroll
  for (int j = 0; j < 4; ++j)
    se[4 * tid + j] = fmaxf(__expf(es[j] - m), 1e-5f);
  __syncthreads();

  float den[4];
#pragma unroll
  for (int j = 0; j < 4; ++j) {
    int k = 4 * tid + j;
    float d = se[k];
    if (k >= 1) d += se[k - 1];
    if (k >= 2) d += se[k - 2];
    if (k >= 3) d += se[k - 3];
    den[j] = d;
  }

  for (int hma = 0; hma < H_MA; ++hma) {
    const float* al = ALPHA + (((size_t)(b * H_MA + hma) * QLEN) + q) * KLEN;
#pragma unroll
    for (int j = 0; j < 4; ++j) {
      int k = 4 * tid + j;
      fr[k] = al[k] / den[j];           // SHARP == 1.0
    }
    __syncthreads();
    bf16* bout = BETA + (((size_t)(b * (H_MA * H_CA) + hma * H_CA + hca) * QLEN) + q) * KLEN;
#pragma unroll
    for (int j = 0; j < 4; ++j) {
      int k = 4 * tid + j;
      float f = fr[k];
      if (k + 1 < KLEN) f += fr[k + 1];
      if (k + 2 < KLEN) f += fr[k + 2];
      if (k + 3 < KLEN) f += fr[k + 3];
      bout[k] = (bf16)(se[k] * f);
    }
    __syncthreads();
  }
}

// ---------------- context: cv[b,q,h*64+d] = sum_k beta[b,h,q,k]*v[b,k,h*64+d]
// V is stored TRANSPOSED: VT[n][b*KLEN + k], n in [0,512). 32x64 tile per wave.

__global__ __launch_bounds__(32)
void context_kernel(const bf16* __restrict__ BETA, const bf16* __restrict__ VT,
                    bf16* __restrict__ CV) {
  int lane = threadIdx.x;
  int bz = blockIdx.z;
  int b = bz >> 3, h = bz & 7;
  int m0 = blockIdx.y * (16 * MT);   // q  (n covers all 64 dims via NT=4)
  const bf16* Ab = BETA + ((size_t)(b * 8 + h) * QLEN) * KLEN;
  const bf16* Vb = VT + (size_t)(h * 64) * (Bsz * KLEN) + (size_t)b * KLEN;
  v8f acc[MT][NT] = {};
  for (int k0 = 0; k0 < KLEN; k0 += 32) {
    if (k0 + 64 < KLEN)
      __builtin_prefetch(Ab + (size_t)(m0 + (lane & 15)) * KLEN + k0 + 64, 0, 3);
    v16bf a[MT], bb[NT];
#pragma unroll
    for (int i = 0; i < MT; ++i)
      a[i] = frag_rowmajor(Ab, KLEN, m0 + 16 * i, k0, lane);
#pragma unroll
    for (int j = 0; j < NT; ++j)
      bb[j] = frag_b_nmajor(Vb, Bsz * KLEN, k0, 16 * j, lane);
#pragma unroll
    for (int i = 0; i < MT; ++i)
#pragma unroll
      for (int j = 0; j < NT; ++j)
        acc[i][j] = wmma_bf16(a[i], bb[j], acc[i][j]);
  }
  int half = lane >> 4;
#pragma unroll
  for (int i = 0; i < MT; ++i)
#pragma unroll
    for (int j = 0; j < NT; ++j) {
      int n = 16 * j + (lane & 15);
#pragma unroll
      for (int v = 0; v < 8; ++v) {
        int mq = m0 + 16 * i + v + (half << 3);
        CV[((size_t)b * QLEN + mq) * ADIM + h * 64 + n] = (bf16)acc[i][j][v];
      }
    }
}

// ---------------------------------------------------------------------------

extern "C" void kernel_launch(void* const* d_in, const int* in_sizes, int n_in,
                              void* d_out, int out_size, void* d_ws, size_t ws_size,
                              hipStream_t stream) {
  (void)in_sizes; (void)n_in; (void)out_size; (void)ws_size;

  const float* key_t = (const float*)d_in[0];
  const float* query = (const float*)d_in[1];
  const float* wk_ma = (const float*)d_in[2];
  const float* bk_ma = (const float*)d_in[3];
  const float* wq_ma = (const float*)d_in[4];
  const float* bq_ma = (const float*)d_in[5];
  const float* r_ptr = (const float*)d_in[6];
  const float* wk_ca = (const float*)d_in[7];
  const float* bk_ca = (const float*)d_in[8];
  const float* wq_ca = (const float*)d_in[9];
  const float* bq_ca = (const float*)d_in[10];
  const float* wv    = (const float*)d_in[11];
  const float* bv    = (const float*)d_in[12];
  const float* wo    = (const float*)d_in[13];
  const float* bo    = (const float*)d_in[14];

  // -------- workspace layout (everything stays resident in the 192MB L2) ----
  char* p = (char*)d_ws;
  auto alloc = [&](size_t bytes) -> char* {
    char* ret = p;
    p += (bytes + 255) & ~(size_t)255;
    return ret;
  };
  const size_t NKEY = (size_t)Bsz * KLEN * ADIM;          // 2M elements
  const size_t NQRY = (size_t)Bsz * QLEN * ADIM;          // 1M elements
  const size_t NW   = (size_t)ADIM * ADIM;                // 256K elements
  const size_t NEMA = (size_t)Bsz * H_MA * QLEN * KLEN;   // 8M
  const size_t NECA = (size_t)Bsz * H_CA * QLEN * KLEN;   // 4M
  const size_t NBETA = (size_t)Bsz * H_MA * H_CA * QLEN * KLEN;  // 16M

  bf16* key_bf  = (bf16*)alloc(NKEY * 2);
  bf16* qry_bf  = (bf16*)alloc(NQRY * 2);
  bf16* wkmaT   = (bf16*)alloc(NW * 2);   // all weights stored [N,K]
  bf16* wqmaT   = (bf16*)alloc(NW * 2);
  bf16* wkcaT   = (bf16*)alloc(NW * 2);
  bf16* wqcaT   = (bf16*)alloc(NW * 2);
  bf16* wvT     = (bf16*)alloc(NW * 2);
  bf16* woT     = (bf16*)alloc(NW * 2);
  bf16* kma_bf  = (bf16*)alloc(NKEY * 2);
  bf16* kca_bf  = (bf16*)alloc(NKEY * 2);
  bf16* vT_bf   = (bf16*)alloc(NKEY * 2); // V projection stored [ADIM, B*KLEN]
  bf16* qma_bf  = (bf16*)alloc(NQRY * 2);
  bf16* qca_bf  = (bf16*)alloc(NQRY * 2);
  float* e_ma   = (float*)alloc(NEMA * 4);
  float* cp     = (float*)alloc(NEMA * 4);
  float* alpha  = (float*)alloc(NEMA * 4);
  float* e_ca   = (float*)alloc(NECA * 4);
  bf16* beta_bf = (bf16*)alloc(NBETA * 2);
  bf16* cv_bf   = (bf16*)alloc(NQRY * 2);

  // -------- 1) activations -> bf16; weights -> bf16 transposed [N,K] ---------
  auto cvt = [&](const float* src, bf16* dst, size_t n) {
    f32_to_bf16_kernel<<<dim3((unsigned)((n + 255) / 256)), 256, 0, stream>>>(src, dst, (int)n);
  };
  auto cvtT = [&](const float* src, bf16* dst) {
    f32_to_bf16_T_kernel<<<dim3((unsigned)((NW + 255) / 256)), 256, 0, stream>>>(src, dst, ADIM, ADIM);
  };
  cvt(key_t, key_bf, NKEY);
  cvt(query, qry_bf, NQRY);
  cvtT(wk_ma, wkmaT);
  cvtT(wq_ma, wqmaT);
  cvtT(wk_ca, wkcaT);
  cvtT(wq_ca, wqcaT);
  cvtT(wv,    wvT);
  cvtT(wo,    woT);

  // -------- 2) projection GEMMs (WMMA bf16, 32x64 tile / wave) ---------------
  const int MKEY = Bsz * KLEN, MQRY = Bsz * QLEN;
  dim3 gK(ADIM / 64, MKEY / 32);   // 8 x 128 waves
  dim3 gQ(ADIM / 64, MQRY / 32);   // 8 x 64 waves
  gemm_bias_kernel<false, false><<<gK, 32, 0, stream>>>(key_bf, ADIM, wkmaT, ADIM, MKEY, bk_ma, kma_bf);
  gemm_bias_kernel<false, false><<<gK, 32, 0, stream>>>(key_bf, ADIM, wkcaT, ADIM, MKEY, bk_ca, kca_bf);
  gemm_bias_kernel<false, true ><<<gK, 32, 0, stream>>>(key_bf, ADIM, wvT,   ADIM, MKEY, bv,    vT_bf);
  gemm_bias_kernel<false, false><<<gQ, 32, 0, stream>>>(qry_bf, ADIM, wqmaT, ADIM, MQRY, bq_ma, qma_bf);
  gemm_bias_kernel<false, false><<<gQ, 32, 0, stream>>>(qry_bf, ADIM, wqcaT, ADIM, MQRY, bq_ca, qca_bf);

  // -------- 3) batched per-head energies (WMMA bf16) -------------------------
  const float inv_scale = 1.0f / SCALEc;
  energy_kernel<<<dim3(KLEN / 64, QLEN / 32, Bsz * H_MA), 32, 0, stream>>>(
      qma_bf, kma_bf, e_ma, H_MA, ADIM / H_MA, inv_scale, r_ptr);
  energy_kernel<<<dim3(KLEN / 64, QLEN / 32, Bsz * H_CA), 32, 0, stream>>>(
      qca_bf, kca_bf, e_ca, H_CA, ADIM / H_CA, inv_scale, nullptr);

  // -------- 4) exclusive cumprod(1-p) per row (parallel over 8192 rows) ------
  cp_kernel<<<dim3(Bsz * H_MA * QLEN), 256, 0, stream>>>(e_ma, cp);

  // -------- 5) sequential MoChA recurrence (16 x 32-wave blocks) -------------
  alpha_scan_kernel<<<dim3(Bsz * H_MA), 1024, 0, stream>>>(e_ma, cp, alpha);

  // -------- 6) fused chunkwise softmax + beta --------------------------------
  beta_kernel<<<dim3(Bsz * H_CA * QLEN), 256, 0, stream>>>(e_ca, alpha, beta_bf);

  // -------- 7) context vectors (WMMA bf16) -----------------------------------
  context_kernel<<<dim3(1, QLEN / 32, Bsz * 8), 32, 0, stream>>>(beta_bf, vT_bf, cv_bf);

  // -------- 8) output projection -> d_out (f32) ------------------------------
  gemm_bias_kernel<true, false><<<gQ, 32, 0, stream>>>(cv_bf, ADIM, woT, ADIM, MQRY, bo, (float*)d_out);
}
